// LSTMForecaster_18202071400535
// MI455X (gfx1250) — matrix operand order repaired and checked by reading, not compiled, not run on hardware
//
#include <hip/hip_runtime.h>
#include <hip/hip_bf16.h>

// ---------------- problem constants ----------------
constexpr int Bb = 512;           // batch
constexpr int Tt = 256;           // time steps
constexpr int In = 32;            // input dim
constexpr int Hh = 128;           // hidden
constexpr int Gg = 4 * Hh;        // gates = 512
constexpr int H2 = Hh / 2;        // fc1 = 64
constexpr int BT = Bb * Tt;       // 131072 rows

// ---------------- vector types ----------------
typedef __attribute__((ext_vector_type(2)))  float        f32x2;
typedef __attribute__((ext_vector_type(4)))  float        f32x4;
typedef __attribute__((ext_vector_type(8)))  float        f32x8;
typedef __attribute__((ext_vector_type(4)))  unsigned int u32x4;
typedef __attribute__((ext_vector_type(4)))  int          i32x4;
typedef __attribute__((ext_vector_type(8)))  int          i32x8;
typedef __attribute__((ext_vector_type(16))) __bf16       bf16x16;

union BFrag {
    bf16x16      v;
    unsigned int u[8];
    u32x4        q[2];
};

// ---------------- helpers ----------------
__device__ __forceinline__ unsigned bf16_rn(float f) {
    union { float f; unsigned u; } a; a.f = f;
    return (a.u + 0x7fffu + ((a.u >> 16) & 1u)) >> 16;
}
__device__ __forceinline__ unsigned bfpack(float lo, float hi) {
    return bf16_rn(lo) | (bf16_rn(hi) << 16);
}
__device__ __forceinline__ float fsig(float x) {
    return 1.0f / (1.0f + __expf(-x));
}
__device__ __forceinline__ float ftanh_f(float x) {
    float e = __expf(-2.0f * x);
    return (1.0f - e) / (1.0f + e);
}

// ---------------------------------------------------------------------
// TDM: async DMA of a 2-D bf16 matrix [dim1 rows x dim0 cols] from global
// into LDS at byte offset ldsOff, inserting 16B of row padding every row
// (pad_amount = 4 dwords, pad_interval = row bytes) -> LDS pitch dim0+8.
// D# layout per CDNA5 ISA 8.3/8.4.  Issue from one wave, then
// s_wait_tensorcnt 0 + workgroup barrier.
// (6-arg builtin form: g0, g1, g2, g3, extra group, cpol)
// ---------------------------------------------------------------------
__device__ __forceinline__ void tdm_load_2d_bf16(const unsigned short* gsrc,
                                                 unsigned ldsOff,
                                                 int dim0, int dim1,
                                                 int padIntervalCode,
                                                 int padAmountCode)
{
    unsigned long long ga = (unsigned long long)(uintptr_t)gsrc;
    u32x4 g0;
    g0[0] = 1u;                                            // count = 1 valid descriptor
    g0[1] = ldsOff;                                        // lds_addr (bytes)
    g0[2] = (unsigned)ga;                                  // global_addr[31:0]
    g0[3] = ((unsigned)(ga >> 32) & 0x01ffffffu)           // global_addr[56:32]
          | (2u << 30);                                    // type = 2 ("image")
    i32x8 g1;
    g1[0] = (int)((1u << 16)                               // data_size = 1 (2 bytes)
          | (1u << 20)                                     // pad_enable
          | ((unsigned)padIntervalCode << 22)              // pad_interval
          | ((unsigned)padAmountCode << 25));              // pad_amount
    g1[1] = (int)(((unsigned)dim0 & 0xffffu) << 16);       // tensor_dim0[15:0] @bits63:48
    g1[2] = (int)(((unsigned)dim0 >> 16)                   // tensor_dim0[31:16]
          | (((unsigned)dim1 & 0xffffu) << 16));           // tensor_dim1[15:0]
    g1[3] = (int)(((unsigned)dim1 >> 16)                   // tensor_dim1[31:16]
          | ((unsigned)dim0 << 16));                       // tile_dim0 @bits127:112
    g1[4] = (int)((unsigned)dim1 & 0xffffu);               // tile_dim1; tile_dim2 = 0
    g1[5] = (int)(unsigned)dim0;                           // tensor_dim0_stride[31:0]
    g1[6] = 0;                                             // stride0[47:32] | stride1[15:0]
    g1[7] = 0;
    i32x4 gz4 = (i32x4)0;                                  // groups 2/3: 2-D tensor
    i32x8 gz8 = (i32x8)0;
    __builtin_amdgcn_tensor_load_to_lds(g0, g1, gz4, gz4, gz8, 0);
}

// =====================================================================
// Kernel 0: f32 -> packed bf16 conversion (weights)
// =====================================================================
__global__ void __launch_bounds__(256)
cvt_bf16_kernel(const float* __restrict__ src, unsigned short* __restrict__ dst, int nPairs)
{
    int i = blockIdx.x * 256 + threadIdx.x;
    if (i < nPairs) {
        f32x2 v = *(const f32x2*)(src + (size_t)i * 2);
        *(unsigned*)(dst + (size_t)i * 2) = bfpack(v.x, v.y);
    }
}

// =====================================================================
// Kernel 1: gate precompute  XG[m, g] = sum_k X[m,k]*W[g,k] + bA[g]+bB[g]
//   W arrives as bf16 in global; staged into LDS by the TDM.
//   X is f32 (layer 0 input) or bf16 (layer 1 = h0seq).
// =====================================================================
template <int K, bool XF32>
__global__ void __launch_bounds__(256)
xg_kernel(const void* __restrict__ Xv,               // [BT, K] f32 or bf16
          const unsigned short* __restrict__ Wbf,    // [Gg, K] bf16
          const float* __restrict__ biasA,           // [Gg]
          const float* __restrict__ biasB,           // [Gg]
          float* __restrict__ XG)                    // [BT, Gg] f32
{
    constexpr int WP = K + 8;                 // LDS pitch (TDM pad = 4 dwords/row)
    constexpr int KT = K / 32;
    constexpr int PAD_INTERVAL = (K == 32) ? 3 : 5;   // 64B / 256B rows
    struct __align__(16) Smem {
        unsigned short sW[Gg * WP];
        float sBias[Gg];
    };
    __shared__ Smem sm;

    const int tid  = threadIdx.x;
    const int lane = tid & 31;
    const int wave = tid >> 5;
    const int lh   = lane & 15;
    const int lg   = lane >> 4;

    for (int idx = tid; idx < Gg; idx += 256)
        sm.sBias[idx] = biasA[idx] + biasB[idx];

    if (tid < 32) {                                   // one wave drives the TDM
        tdm_load_2d_bf16(Wbf, 0u /*offset of sW*/, K, Gg, PAD_INTERVAL, 3);
        __builtin_amdgcn_s_wait_tensorcnt(0);
    }
    __syncthreads();

    // A fragments for this wave's 16 rows (reused across all 32 n-tiles)
    const int row0 = blockIdx.x * 128 + wave * 16;
    BFrag afr[KT];
    if constexpr (XF32) {
        const float* rp = (const float*)Xv + (size_t)(row0 + lh) * K;
#pragma unroll
        for (int kt = 0; kt < KT; ++kt) {
            const int kb = kt * 32 + lg * 8;          // A: lane-half -> +0 / +8
            f32x4 f0 = *(const f32x4*)(rp + kb);
            f32x4 f1 = *(const f32x4*)(rp + kb + 4);
            f32x4 f2 = *(const f32x4*)(rp + kb + 16);
            f32x4 f3 = *(const f32x4*)(rp + kb + 20);
            afr[kt].u[0] = bfpack(f0.x, f0.y); afr[kt].u[1] = bfpack(f0.z, f0.w);
            afr[kt].u[2] = bfpack(f1.x, f1.y); afr[kt].u[3] = bfpack(f1.z, f1.w);
            afr[kt].u[4] = bfpack(f2.x, f2.y); afr[kt].u[5] = bfpack(f2.z, f2.w);
            afr[kt].u[6] = bfpack(f3.x, f3.y); afr[kt].u[7] = bfpack(f3.z, f3.w);
        }
    } else {
        const unsigned short* rp = (const unsigned short*)Xv + (size_t)(row0 + lh) * K;
#pragma unroll
        for (int kt = 0; kt < KT; ++kt) {
            const int kb = kt * 32 + lg * 8;
            afr[kt].q[0] = *(const u32x4*)(rp + kb);
            afr[kt].q[1] = *(const u32x4*)(rp + kb + 16);
        }
    }

    for (int nt = 0; nt < Gg / 16; ++nt) {
        f32x8 acc = {};
        const int n = nt * 16 + lh;                   // B: N = lane%16
#pragma unroll
        for (int kt = 0; kt < KT; ++kt) {
            BFrag bfr;
            const unsigned short* wp = sm.sW + n * WP + kt * 32 + lg * 8;
            bfr.q[0] = *(const u32x4*)(wp);
            bfr.q[1] = *(const u32x4*)(wp + 16);
            acc = __builtin_amdgcn_wmma_f32_16x16x32_bf16(
                false, afr[kt].v, false, bfr.v, (short)0, acc, false, false);
        }
        const float bias = sm.sBias[n];
        float* dst = XG + (size_t)(row0 + 8 * lg) * Gg + n;
#pragma unroll
        for (int r = 0; r < 8; ++r)
            dst[(size_t)r * Gg] = acc[r] + bias;
    }
}

// =====================================================================
// Kernel 2: LSTM scan.  Block owns 32 batch rows for all T steps.
//   W_hh staged into LDS by TDM (bf16).  h double-buffered in LDS (bf16).
//   Cell state in accumulator VGPRs.  K=128 -> 4 WMMA per gate tile.
// =====================================================================
__global__ void __launch_bounds__(256)
lstm_scan_kernel(const float* __restrict__ XG,            // [BT, Gg], row m = b*T + t
                 const unsigned short* __restrict__ WhhBf,// [Gg, Hh] bf16
                 unsigned short* __restrict__ HseqOut,    // [BT, Hh] bf16 or nullptr
                 float* __restrict__ HlastOut)            // [Bb, Hh] f32 or nullptr
{
    constexpr int BTILE = 32;
    constexpr int WP    = Hh + 8;             // 136 (matches TDM padding)
    constexpr int SHP   = Hh + 8;
    struct __align__(16) Smem {
        unsigned short sW[Gg * WP];           // 139264 B  (must be first: TDM target @0)
        unsigned short sh[2][BTILE * SHP];    //  17408 B
    };
    __shared__ Smem sm;

    const int tid   = threadIdx.x;
    const int lane  = tid & 31;
    const int wave  = tid >> 5;
    const int lh    = lane & 15;
    const int lg    = lane >> 4;
    const int bbase = blockIdx.x * BTILE;

    // h_0 = 0
    for (int idx = tid; idx < 2 * BTILE * SHP; idx += 256) sm.sh[0][idx] = 0;

    if (tid < 32) {                            // one wave drives the TDM
        tdm_load_2d_bf16(WhhBf, 0u, Hh, Gg, 5, 3);
        __builtin_amdgcn_s_wait_tensorcnt(0);
    }

    f32x8 cst[2];
    cst[0] = (f32x8)0.0f;
    cst[1] = (f32x8)0.0f;

    for (int t = 0; t < Tt; ++t) {
        __syncthreads();                       // h_t (and at t=0, sW/zeroes) visible
        const unsigned short* hin  = sm.sh[t & 1];
        unsigned short*       hout = sm.sh[(t + 1) & 1];

        // A fragments from h_t: both row-tiles, 4 k-tiles each
        BFrag afr[2][4];
#pragma unroll
        for (int rt = 0; rt < 2; ++rt) {
            const unsigned short* rp = hin + (rt * 16 + lh) * SHP;
#pragma unroll
            for (int kt = 0; kt < 4; ++kt) {
                const int kb = kt * 32 + lg * 8;
                afr[rt][kt].q[0] = *(const u32x4*)(rp + kb);
                afr[rt][kt].q[1] = *(const u32x4*)(rp + kb + 16);
            }
        }

#pragma unroll
        for (int p = 0; p < 2; ++p) {
            const int pair = wave * 2 + p;
            const int rt   = pair & 1;         // row tile 0/1
            const int hc   = pair >> 1;        // hidden col tile 0..7
            const int b0   = bbase + rt * 16 + 8 * lg;

            // init gate accumulators from precomputed xg (read-once: NT)
            f32x8 acc[4];
#pragma unroll
            for (int q = 0; q < 4; ++q) {
                const int col = q * Hh + hc * 16 + lh;
                const float* src = XG + ((size_t)b0 * Tt + t) * Gg + col;
#pragma unroll
                for (int r = 0; r < 8; ++r)
                    acc[q][r] = __builtin_nontemporal_load(src + (size_t)r * Tt * Gg);
                if (t + 1 < Tt)
                    __builtin_prefetch(src + Gg, 0, 1);
            }

            // gates += h_t @ W_hh^T
#pragma unroll
            for (int kt = 0; kt < 4; ++kt) {
#pragma unroll
                for (int q = 0; q < 4; ++q) {
                    const int n = q * Hh + hc * 16 + lh;
                    const unsigned short* wp = sm.sW + n * WP + kt * 32 + lg * 8;
                    BFrag bfr;
                    bfr.q[0] = *(const u32x4*)(wp);
                    bfr.q[1] = *(const u32x4*)(wp + 16);
                    acc[q] = __builtin_amdgcn_wmma_f32_16x16x32_bf16(
                        false, afr[rt][kt].v, false, bfr.v, (short)0, acc[q], false, false);
                }
            }

            // elementwise LSTM cell update (gate order: i, f, g, o)
            f32x8 hval;
            unsigned short hbf[8];
#pragma unroll
            for (int r = 0; r < 8; ++r) {
                float ig = fsig(acc[0][r]);
                float fg = fsig(acc[1][r]);
                float gg = ftanh_f(acc[2][r]);
                float og = fsig(acc[3][r]);
                float cn = fg * cst[p][r] + ig * gg;
                cst[p][r] = cn;
                hval[r] = og * ftanh_f(cn);
                hbf[r]  = (unsigned short)bf16_rn(hval[r]);
            }

            // write h_{t+1}: bf16 -> LDS (next step's A) and bf16 -> global seq
            const int ncol = hc * 16 + lh;
#pragma unroll
            for (int r = 0; r < 8; ++r)
                hout[(rt * 16 + r + 8 * lg) * SHP + ncol] = hbf[r];

            if (HseqOut) {
                unsigned short* dst = HseqOut + ((size_t)b0 * Tt + t) * Hh + ncol;
#pragma unroll
                for (int r = 0; r < 8; ++r)
                    dst[(size_t)r * Tt * Hh] = hbf[r];
            }
            if (HlastOut && t == Tt - 1) {
                float* dst = HlastOut + (size_t)b0 * Hh + ncol;
#pragma unroll
                for (int r = 0; r < 8; ++r)
                    dst[(size_t)r * Hh] = hval[r];
            }
        }
    }
}

// =====================================================================
// Kernel 3: FC head.  out[b] = fc2_w . relu(fc1_w @ h_last[b] + fc1_b) + fc2_b
// =====================================================================
__global__ void __launch_bounds__(64)
fc_head_kernel(const float* __restrict__ Hlast,  // [Bb, Hh]
               const float* __restrict__ W1,     // [H2, Hh]
               const float* __restrict__ B1,     // [H2]
               const float* __restrict__ W2,     // [1, H2]
               const float* __restrict__ B2,     // [1]
               float* __restrict__ Out)          // [Bb]
{
    __shared__ float red[H2];
    const int b = blockIdx.x;
    const int j = threadIdx.x;
    const float* h = Hlast + (size_t)b * Hh;
    const float* w = W1 + (size_t)j * Hh;
    float acc = B1[j];
#pragma unroll 4
    for (int k = 0; k < Hh; ++k) acc += h[k] * w[k];
    red[j] = fmaxf(acc, 0.0f) * W2[j];
    __syncthreads();
#pragma unroll
    for (int s = 32; s > 0; s >>= 1) {
        if (j < s) red[j] += red[j + s];
        __syncthreads();
    }
    if (j == 0) Out[b] = red[0] + B2[0];
}

// =====================================================================
extern "C" void kernel_launch(void* const* d_in, const int* in_sizes, int n_in,
                              void* d_out, int out_size, void* d_ws, size_t ws_size,
                              hipStream_t stream)
{
    const float* x     = (const float*)d_in[0];
    const float* w_ih0 = (const float*)d_in[1];
    const float* w_hh0 = (const float*)d_in[2];
    const float* b_ih0 = (const float*)d_in[3];
    const float* b_hh0 = (const float*)d_in[4];
    const float* w_ih1 = (const float*)d_in[5];
    const float* w_hh1 = (const float*)d_in[6];
    const float* b_ih1 = (const float*)d_in[7];
    const float* b_hh1 = (const float*)d_in[8];
    const float* fc1_w = (const float*)d_in[9];
    const float* fc1_b = (const float*)d_in[10];
    const float* fc2_w = (const float*)d_in[11];
    const float* fc2_b = (const float*)d_in[12];
    float* out = (float*)d_out;

    // -------- workspace layout (byte offsets, 16B-aligned regions) --------
    char* ws = (char*)d_ws;
    float*          xg     = (float*)ws;                               // 256 MB
    unsigned short* h0seq  = (unsigned short*)(ws + (size_t)BT * Gg * 4);      // 32 MB bf16
    float*          h1last = (float*)((char*)h0seq + (size_t)BT * Hh * 2);     // 256 KB
    unsigned short* wih0b  = (unsigned short*)((char*)h1last + (size_t)Bb * Hh * 4);
    unsigned short* whh0b  = wih0b + (size_t)Gg * In;
    unsigned short* wih1b  = whh0b + (size_t)Gg * Hh;
    unsigned short* whh1b  = wih1b + (size_t)Gg * Hh;

    // -------- weight f32 -> bf16 (consumed by the TDM loads) --------
    cvt_bf16_kernel<<<(Gg * In / 2 + 255) / 256, 256, 0, stream>>>(w_ih0, wih0b, Gg * In / 2);
    cvt_bf16_kernel<<<(Gg * Hh / 2 + 255) / 256, 256, 0, stream>>>(w_hh0, whh0b, Gg * Hh / 2);
    cvt_bf16_kernel<<<(Gg * Hh / 2 + 255) / 256, 256, 0, stream>>>(w_ih1, wih1b, Gg * Hh / 2);
    cvt_bf16_kernel<<<(Gg * Hh / 2 + 255) / 256, 256, 0, stream>>>(w_hh1, whh1b, Gg * Hh / 2);

    // layer 0
    xg_kernel<In, true><<<BT / 128, 256, 0, stream>>>(x, wih0b, b_ih0, b_hh0, xg);
    lstm_scan_kernel<<<Bb / 32, 256, 0, stream>>>(xg, whh0b, h0seq, nullptr);
    // layer 1
    xg_kernel<Hh, false><<<BT / 128, 256, 0, stream>>>(h0seq, wih1b, b_ih1, b_hh1, xg);
    lstm_scan_kernel<<<Bb / 32, 256, 0, stream>>>(xg, whh1b, nullptr, h1last);
    // head
    fc_head_kernel<<<Bb, 64, 0, stream>>>(h1last, fc1_w, fc1_b, fc2_w, fc2_b, out);
}
